// HGT_6305011991205
// MI455X (gfx1250) — matrix operand order repaired and checked by reading, not compiled, not run on hardware
//
#include <hip/hip_runtime.h>
#include <hip/hip_bf16.h>
#include <math.h>

// ---------------------------------------------------------------------------
// HGT on MI455X (gfx1250). Dense GEMMs via v_wmma_f32_16x16x32_bf16 (fp32
// accumulate), graph gather/softmax/scatter in fp32 with atomics.
// ---------------------------------------------------------------------------

typedef __attribute__((ext_vector_type(8)))  float  v8f;
typedef __attribute__((ext_vector_type(16))) __bf16 v16bf;

#define WG 256

__device__ __forceinline__ unsigned short f32_to_bf16(float f) {
  unsigned int u = __float_as_uint(f);
  u += 0x7fffu + ((u >> 16) & 1u);            // round-to-nearest-even
  return (unsigned short)(u >> 16);
}
// order-preserving float -> uint map for atomicMax-based segment max
__device__ __forceinline__ unsigned int fmapu(float f) {
  unsigned int b = __float_as_uint(f);
  return (b & 0x80000000u) ? ~b : (b | 0x80000000u);
}
__device__ __forceinline__ float funmap0(unsigned int u) {
  if (u == 0u) return 0.0f;                    // untouched segment
  unsigned int b = (u & 0x80000000u) ? (u & 0x7fffffffu) : ~u;
  return __uint_as_float(b);
}

union Frag16 { uint4 u[2]; v16bf v; };

// ---------------------------------------------------------------------------
// Utility kernels
// ---------------------------------------------------------------------------
__global__ void k_fill_u32(unsigned int* __restrict__ p, unsigned int v, int n) {
  int i = blockIdx.x * WG + threadIdx.x;
  if (i < n) p[i] = v;
}

__global__ void k_cvt_bf16(const float* __restrict__ s,
                           unsigned short* __restrict__ d, int n) {
  int i = blockIdx.x * WG + threadIdx.x;
  if (i < n) d[i] = f32_to_bf16(s[i]);
}

__global__ void k_gelu_cvt_bf16(const float* __restrict__ s,
                                unsigned short* __restrict__ d, int n) {
  int i = blockIdx.x * WG + threadIdx.x;
  if (i < n) {
    float x = s[i];
    float g = 0.5f * x * (1.0f + erff(x * 0.70710678118654752f));  // exact gelu
    d[i] = f32_to_bf16(g);
  }
}

// expand rel[8,16,16] (h, d_in, d_out) into a 128x128 block-diagonal bf16 W
__global__ void k_blockdiag(const float* __restrict__ rel,
                            unsigned short* __restrict__ w) {
  int i = blockIdx.x * WG + threadIdx.x;      // 128*128 = 16384 threads
  if (i >= 128 * 128) return;
  int r = i >> 7, c = i & 127;
  int hr = r >> 4, hc = c >> 4;
  float v = (hr == hc) ? rel[hr * 256 + (r & 15) * 16 + (c & 15)] : 0.0f;
  w[i] = f32_to_bf16(v);
}

// ---------------------------------------------------------------------------
// WMMA GEMM: C[M,N] = epilogue(A[M,K](bf16) @ B[K,N](bf16) + bias)
//  mode 0: none   mode 1: relu   mode 2: sigmoid-skip blend with Xprev
//  Optional bf16 mirror of C in Cbf.
//  Block = 256 threads = 8 waves; wave w owns rows [wave*16,+16) of a 128-row
//  tile and NT accumulators of 16 cols each. K stepped by 32.
// ---------------------------------------------------------------------------
template <int NT>
__global__ __launch_bounds__(WG) void k_gemm(
    const unsigned short* __restrict__ A, const unsigned short* __restrict__ B,
    const float* __restrict__ bias, float* __restrict__ C,
    unsigned short* __restrict__ Cbf, const float* __restrict__ skipp,
    const float* __restrict__ Xprev, int M, int N, int K, int mode) {
  constexpr int BN = NT * 16;
  __shared__ alignas(16) unsigned short ldsA[128 * 40];  // padded stride: no conflicts
  __shared__ alignas(16) unsigned short ldsB[BN * 40];   // stored [col][k]

  const int tid = threadIdx.x;
  const int wave = tid >> 5;
  const int lane = tid & 31;
  const int rowBase = blockIdx.x * 128;
  const int colBase = blockIdx.y * BN;

  v8f acc[NT];
#pragma unroll
  for (int g = 0; g < NT; ++g)
#pragma unroll
    for (int r = 0; r < 8; ++r) acc[g][r] = 0.0f;

  // cooperative A load: thread -> (row = tid/2, 16-col half = tid&1)
  const int arow = tid >> 1;
  const int ahalf = (tid & 1) * 16;
  const bool aValid = (rowBase + arow) < M;
  const unsigned short* aPtr = A + (long)(rowBase + arow) * K + ahalf;

  // WMMA operand layouts (ISA 7.12.2, 16-bit, wave32)
  const int awrow = wave * 16 + (lane & 15);
  const int akoff = (lane < 16) ? 0 : 8;     // A: lanes 0-15 K 0-7/16-23; 16-31 K 8-15/24-31
  const int bkoff = (lane < 16) ? 0 : 8;     // B: lanes 0-15 K 0-7/16-23; 16-31 K 8-15/24-31

  for (int k0 = 0; k0 < K; k0 += 32) {
    uint4 a0 = make_uint4(0, 0, 0, 0), a1 = make_uint4(0, 0, 0, 0);
    if (aValid) {
      const uint4* p = (const uint4*)(aPtr + k0);
      a0 = p[0];
      a1 = p[1];
      if (k0 + 32 < K) __builtin_prefetch(aPtr + k0 + 32, 0, 1);  // global_prefetch
    }
    *(uint4*)&ldsA[arow * 40 + ahalf] = a0;
    *(uint4*)&ldsA[arow * 40 + ahalf + 8] = a1;

    for (int g = tid; g < 32 * (BN / 8); g += WG) {
      int krow = g / (BN / 8);
      int cg = g % (BN / 8);
      uint4 bv = *(const uint4*)(B + (long)(k0 + krow) * N + colBase + cg * 8);
      const unsigned short* pv = (const unsigned short*)&bv;
#pragma unroll
      for (int j = 0; j < 8; ++j) ldsB[(cg * 8 + j) * 40 + krow] = pv[j];
    }
    __syncthreads();

    Frag16 af;
    af.u[0] = *(const uint4*)&ldsA[awrow * 40 + akoff];
    af.u[1] = *(const uint4*)&ldsA[awrow * 40 + akoff + 16];

#pragma unroll
    for (int g = 0; g < NT; ++g) {
      Frag16 bf;
      const int bcol = g * 16 + (lane & 15);
      bf.u[0] = *(const uint4*)&ldsB[bcol * 40 + bkoff];
      bf.u[1] = *(const uint4*)&ldsB[bcol * 40 + bkoff + 16];
      acc[g] = __builtin_amdgcn_wmma_f32_16x16x32_bf16(
          false, af.v, false, bf.v, (short)0, acc[g], false, false);
    }
    __syncthreads();
  }

  const int tileRow = rowBase + wave * 16;
  if (tileRow >= M) return;  // M is a multiple of 16: whole wave-tile in/out
  const int wrow0 = tileRow + ((lane < 16) ? 0 : 8);  // C layout (ISA 7.12.2)
  float aS = 1.0f, bS = 0.0f;
  if (mode == 2) {
    aS = 1.0f / (1.0f + expf(-skipp[0]));
    bS = 1.0f - aS;
  }
#pragma unroll
  for (int g = 0; g < NT; ++g) {
    const int colg = colBase + g * 16 + (lane & 15);
    const float bb = bias ? bias[colg] : 0.0f;
#pragma unroll
    for (int r = 0; r < 8; ++r) {
      const long idx = (long)(wrow0 + r) * N + colg;
      float v = acc[g][r] + bb;
      if (mode == 1)
        v = fmaxf(v, 0.0f);
      else if (mode == 2)
        v = aS * v + bS * Xprev[idx];
      C[idx] = v;
      if (Cbf) Cbf[idx] = f32_to_bf16(v);
    }
  }
}

// ---------------------------------------------------------------------------
// Edge phase. Thread per (edge, head). scores buffer is [E,8].
// ---------------------------------------------------------------------------
__global__ void k_scores(const int* __restrict__ eidx, int E,
                         const float* __restrict__ kt,
                         const float* __restrict__ q,
                         const float* __restrict__ prel,
                         float* __restrict__ sc,
                         unsigned int* __restrict__ smax) {
  int i = blockIdx.x * WG + threadIdx.x;
  if (i >= E * 8) return;
  int h = i & 7, e = i >> 3;
  int s = eidx[e], d = eidx[E + e];
  const float* qp = q + (long)d * 128 + h * 16;
  const float* kp = kt + (long)s * 128 + h * 16;
  float a = 0.0f;
#pragma unroll
  for (int j = 0; j < 16; ++j) a += qp[j] * kp[j];
  a *= prel[h] * 0.25f;  // 1/sqrt(DH), DH=16
  sc[i] = a;
  atomicMax(&smax[(long)d * 8 + h], fmapu(a));
}

__global__ void k_expsum(const int* __restrict__ eidx, int E,
                         float* __restrict__ sc,
                         const unsigned int* __restrict__ smax,
                         float* __restrict__ ssum) {
  int i = blockIdx.x * WG + threadIdx.x;
  if (i >= E * 8) return;
  int h = i & 7, e = i >> 3;
  int d = eidx[E + e];
  float m = funmap0(smax[(long)d * 8 + h]);
  float ex = expf(sc[i] - m);
  sc[i] = ex;
  atomicAdd(&ssum[(long)d * 8 + h], ex);
}

__global__ void k_scatter(const int* __restrict__ eidx, int E,
                          const float* __restrict__ sc,
                          const float* __restrict__ ssum,
                          const float* __restrict__ vt,
                          float* __restrict__ agg) {
  int i = blockIdx.x * WG + threadIdx.x;
  if (i >= E * 8) return;
  int h = i & 7, e = i >> 3;
  int s = eidx[e], d = eidx[E + e];
  float w = sc[i] / (ssum[(long)d * 8 + h] + 1e-16f);
  const float* vp = vt + (long)s * 128 + h * 16;
  float* ap = agg + (long)d * 128 + h * 16;
#pragma unroll
  for (int j = 0; j < 16; ++j) atomicAdd(ap + j, vp[j] * w);
}

// ---------------------------------------------------------------------------
// Host orchestration
// ---------------------------------------------------------------------------
extern "C" void kernel_launch(void* const* d_in, const int* in_sizes, int n_in,
                              void* d_out, int out_size, void* d_ws,
                              size_t ws_size, hipStream_t stream) {
  const int NA = 50000, E = 500000;
  (void)in_sizes; (void)n_in; (void)out_size;

  const float* xA_in = (const float*)d_in[0];
  const float* xP_in = (const float*)d_in[1];
  const int* eW = (const int*)d_in[2];
  const int* eR = (const int*)d_in[3];
  auto P = [&](int i) { return (const float*)d_in[i]; };

  // bump allocator over workspace
  char* wsb = (char*)d_ws;
  size_t off = 0;
  auto alloc = [&](size_t bytes) -> void* {
    off = (off + 255) & ~(size_t)255;
    void* p = wsb + off;
    off += bytes;
    return p;
  };
  auto fU16 = [&](size_t n) { return (unsigned short*)alloc(n * 2); };
  auto fF32 = [&](size_t n) { return (float*)alloc(n * 4); };

  // bf16 weights
  unsigned short* WinA = fU16(256 * 128);
  unsigned short* WinP = fU16(256 * 128);
  unsigned short *Wk[2][2], *Wq[2][2], *Wv[2][2], *Wal[2][2], *BDa[2][2], *BDm[2][2];
  for (int l = 0; l < 2; ++l)
    for (int t = 0; t < 2; ++t) {
      Wk[l][t] = fU16(128 * 128); Wq[l][t] = fU16(128 * 128);
      Wv[l][t] = fU16(128 * 128); Wal[l][t] = fU16(128 * 128);
      BDa[l][t] = fU16(128 * 128); BDm[l][t] = fU16(128 * 128);
    }
  unsigned short* Wout = fU16(128 * 16);

  const size_t NF = (size_t)NA * 128;
  unsigned short* xinAbf = fU16((size_t)NA * 256);
  unsigned short* xinPbf = fU16((size_t)NA * 256);
  unsigned short* xbf[2] = {fU16(NF), fU16(NF)};
  unsigned short* kvbf[2] = {fU16(NF), fU16(NF)};
  unsigned short* aggbf = fU16(NF);

  float* x0[2] = {fF32(NF), fF32(NF)};
  float* x1[2] = {fF32(NF), fF32(NF)};
  float* qf[2] = {fF32(NF), fF32(NF)};
  float* tmp = fF32(NF);
  float* kt[2] = {fF32(NF), fF32(NF)};
  float* vt[2] = {fF32(NF), fF32(NF)};
  float* agg[2] = {fF32(NF), fF32(NF)};
  float* scores = fF32((size_t)E * 8);
  float* segsum = fF32((size_t)NA * 8);
  unsigned int* segmax = (unsigned int*)alloc((size_t)NA * 8 * 4);

  if (off > ws_size) return;  // insufficient scratch: bail (no UB)

  auto cvt = [&](const float* s, unsigned short* d, long n) {
    k_cvt_bf16<<<dim3((unsigned)((n + WG - 1) / WG)), dim3(WG), 0, stream>>>(s, d, (int)n);
  };
  auto fillz = [&](void* p, long n) {  // n u32 words
    k_fill_u32<<<dim3((unsigned)((n + WG - 1) / WG)), dim3(WG), 0, stream>>>(
        (unsigned int*)p, 0u, (int)n);
  };
  auto gemm8 = [&](const unsigned short* Ab, const unsigned short* Bb,
                   const float* bias, float* C, unsigned short* Cbf,
                   const float* sk, const float* xp, int M, int N, int K, int mode) {
    k_gemm<8><<<dim3((M + 127) / 128, N / 128), dim3(WG), 0, stream>>>(
        Ab, Bb, bias, C, Cbf, sk, xp, M, N, K, mode);
  };

  // ---- weight prep ----
  cvt(P(4), WinA, 256 * 128);
  cvt(P(6), WinP, 256 * 128);
  for (int l = 0; l < 2; ++l) {
    int b8 = 8 + 24 * l;
    cvt(P(b8 + 0), Wk[l][0], 16384);  cvt(P(b8 + 2), Wk[l][1], 16384);
    cvt(P(b8 + 4), Wq[l][0], 16384);  cvt(P(b8 + 6), Wq[l][1], 16384);
    cvt(P(b8 + 8), Wv[l][0], 16384);  cvt(P(b8 + 10), Wv[l][1], 16384);
    cvt(P(b8 + 12), Wal[l][0], 16384); cvt(P(b8 + 14), Wal[l][1], 16384);
    // edge-type index: 0=writes(src=author), 1=rev(src=paper)
    k_blockdiag<<<64, WG, 0, stream>>>(P(b8 + 18), BDa[l][0]);
    k_blockdiag<<<64, WG, 0, stream>>>(P(b8 + 19), BDa[l][1]);
    k_blockdiag<<<64, WG, 0, stream>>>(P(b8 + 20), BDm[l][0]);
    k_blockdiag<<<64, WG, 0, stream>>>(P(b8 + 21), BDm[l][1]);
  }
  cvt(P(56), Wout, 128 * 16);

  // ---- input projection + relu (fused bf16 mirror) ----
  cvt(xA_in, xinAbf, (long)NA * 256);
  cvt(xP_in, xinPbf, (long)NA * 256);
  gemm8(xinAbf, WinA, P(5), x0[0], xbf[0], nullptr, nullptr, NA, 128, 256, 1);
  gemm8(xinPbf, WinP, P(7), x0[1], xbf[1], nullptr, nullptr, NA, 128, 256, 1);

  float* xc[2] = {x0[0], x0[1]};
  float* xn[2] = {x1[0], x1[1]};

  for (int l = 0; l < 2; ++l) {
    int b8 = 8 + 24 * l;
    // Q/K/V projections + relation transforms (block-diagonal GEMMs)
    for (int t = 0; t < 2; ++t) {  // t: 0=author, 1=paper; src of edge-type t
      gemm8(xbf[t], Wq[l][t], P(b8 + 5 + 2 * t), qf[t], nullptr, nullptr, nullptr,
            NA, 128, 128, 0);
      gemm8(xbf[t], Wk[l][t], P(b8 + 1 + 2 * t), tmp, kvbf[t], nullptr, nullptr,
            NA, 128, 128, 0);
      gemm8(kvbf[t], BDa[l][t], nullptr, kt[t], nullptr, nullptr, nullptr,
            NA, 128, 128, 0);
      gemm8(xbf[t], Wv[l][t], P(b8 + 9 + 2 * t), tmp, kvbf[t], nullptr, nullptr,
            NA, 128, 128, 0);
      gemm8(kvbf[t], BDm[l][t], nullptr, vt[t], nullptr, nullptr, nullptr,
            NA, 128, 128, 0);
    }
    // edge attention + scatter
    fillz(agg[0], (long)NA * 128);
    fillz(agg[1], (long)NA * 128);
    for (int et = 0; et < 2; ++et) {
      const int* ei = (et == 0) ? eW : eR;
      int st = et;        // src node type
      int dt = 1 - et;    // dst node type
      const float* prel = P(b8 + 22 + et);
      fillz(segmax, (long)NA * 8);
      fillz(segsum, (long)NA * 8);
      int nb = (E * 8 + WG - 1) / WG;
      k_scores<<<nb, WG, 0, stream>>>(ei, E, kt[st], qf[dt], prel, scores, segmax);
      k_expsum<<<nb, WG, 0, stream>>>(ei, E, scores, segmax, segsum);
      k_scatter<<<nb, WG, 0, stream>>>(ei, E, scores, segsum, vt[st], agg[dt]);
    }
    // gelu -> a_lin -> sigmoid-skip blend (fused), bf16 mirror for next layer
    for (int t = 0; t < 2; ++t) {
      k_gelu_cvt_bf16<<<dim3((unsigned)((NF + WG - 1) / WG)), dim3(WG), 0, stream>>>(
          agg[t], aggbf, (int)NF);
      gemm8(aggbf, Wal[l][t], P(b8 + 13 + 2 * t), xn[t], xbf[t], P(b8 + 16 + t),
            xc[t], NA, 128, 128, 2);
    }
    for (int t = 0; t < 2; ++t) { float* s = xc[t]; xc[t] = xn[t]; xn[t] = s; }
  }

  // ---- output head: paper logits [50000,16] ----
  k_gemm<1><<<dim3((NA + 127) / 128, 1), dim3(WG), 0, stream>>>(
      xbf[1], Wout, P(57), (float*)d_out, nullptr, nullptr, nullptr,
      NA, 16, 128, 0);
}
